// NeuralRoughSimulator_63290638073861
// MI455X (gfx1250) — matrix operand order repaired and checked by reading, not compiled, not run on hardware
//
#include <hip/hip_runtime.h>

typedef __attribute__((ext_vector_type(16))) _Float16 v16h;
typedef __attribute__((ext_vector_type(8)))  float    v8f;

#define SIGD 15

#ifndef __has_builtin
#define __has_builtin(x) 0
#endif

__device__ __forceinline__ float fast_tanh(float x) {
#if __has_builtin(__builtin_amdgcn_tanhf)
  return __builtin_amdgcn_tanhf(x);            // v_tanh_f32 (gfx1250 TRANS)
#else
  // tanh(x) = 1 - 2/(exp(2x)+1), via v_exp_f32 + v_rcp_f32 (TRANS, co-executes)
  float e = __builtin_amdgcn_exp2f(x * 2.885390081777927f); // 2*log2(e)
  return 1.0f - 2.0f * __builtin_amdgcn_rcpf(e + 1.0f);
#endif
}

__device__ __forceinline__ float fast_exp(float x) {
  return __builtin_amdgcn_exp2f(x * 1.4426950408889634f);
}
__device__ __forceinline__ float fast_sigmoid(float x) {
  return __builtin_amdgcn_rcpf(1.0f + __builtin_amdgcn_exp2f(-x * 1.4426950408889634f));
}
__device__ __forceinline__ float fast_log(float x) {
  return __builtin_amdgcn_logf(x) * 0.6931471805599453f;
}

__device__ __forceinline__ v8f wmma_f16(v16h a, v16h b, v8f c) {
  // D = A(16x32 f16) x B(32x16 f16) + C(16x16 f32)
  return __builtin_amdgcn_wmma_f32_16x16x32_f16(false, a, false, b, (short)0, c,
                                                false, false);
}

// One wave32 per 16 paths. Transposed formulation:
//   D(32x16) = W2(32x32, as two 16x32 A tiles) x H1^T(32x16, B operand) + b2 (C operand)
// so layer-1 outputs are produced directly in B-layout and layer-3 is an
// in-lane 16-dot + one shfl_xor(16).
__global__ __launch_bounds__(32) void neural_rough_sde_kernel(
    const float* __restrict__ init_var,
    const float* __restrict__ signatures,
    const float* __restrict__ brownian,
    const float* __restrict__ dt_p,
    const float* __restrict__ dr_w1, const float* __restrict__ dr_b1,
    const float* __restrict__ dr_w2, const float* __restrict__ dr_b2,
    const float* __restrict__ dr_w3, const float* __restrict__ dr_b3,
    const float* __restrict__ df_w1, const float* __restrict__ df_b1,
    const float* __restrict__ df_w2, const float* __restrict__ df_b2,
    const float* __restrict__ df_w3, const float* __restrict__ df_b3,
    float* __restrict__ out, int B, int T) {
  const int lane = threadIdx.x & 31;
  const int half = lane >> 4;   // which half-wave
  const int m    = lane & 15;   // path-in-tile (N index)
  const int p    = blockIdx.x * 16 + m;

  const float dt = dt_p[0];

  // ---- setup: precompute signature part of layer 1 (per-lane k ownership
  //      matches this kernel's contiguous A/B k order: lane<16 -> k=0..15, lane>=16 -> 16..31)
  float sig[SIGD];
  #pragma unroll
  for (int i = 0; i < SIGD; ++i) sig[i] = signatures[p * SIGD + i];

  float pre_dr[16], w1v_dr[16], pre_df[16], w1v_df[16];
  #pragma unroll
  for (int j = 0; j < 16; ++j) {
    const int k = half * 16 + j;
    float adr = dr_b1[k], adf = df_b1[k];
    #pragma unroll
    for (int i = 0; i < SIGD; ++i) {
      adr += sig[i] * dr_w1[k * 16 + i];
      adf += sig[i] * df_w1[k * 16 + i];
    }
    pre_dr[j] = adr; w1v_dr[j] = dr_w1[k * 16 + SIGD];
    pre_df[j] = adf; w1v_df[j] = df_w1[k * 16 + SIGD];
  }

  // ---- W2 as two A-layout 16x32 f16 tiles per MLP (rows 0..15 and 16..31).
  // Contiguous k order (not AMD's table order): lanes0-15 hold K=0..15 (e=0..15);
  // lanes16-31 hold K=16..31 (matches B k ownership).
  v16h A0dr, A1dr, A0df, A1df;
  #pragma unroll
  for (int e = 0; e < 16; ++e) {
    const int K = e + 16 * half;
    A0dr[e] = (_Float16)dr_w2[m * 32 + K];
    A1dr[e] = (_Float16)dr_w2[(16 + m) * 32 + K];
    A0df[e] = (_Float16)df_w2[m * 32 + K];
    A1df[e] = (_Float16)df_w2[(16 + m) * 32 + K];
  }

  // ---- b2 preloaded in C/D layout (VGPR r: M = 8*half + r), w3 likewise.
  v8f Cdr0, Cdr1, Cdf0, Cdf1;
  float w3aDr[8], w3bDr[8], w3aDf[8], w3bDf[8];
  #pragma unroll
  for (int r = 0; r < 8; ++r) {
    const int M0 = 8 * half + r;
    const int M1 = 16 + M0;
    Cdr0[r] = dr_b2[M0]; Cdr1[r] = dr_b2[M1];
    Cdf0[r] = df_b2[M0]; Cdf1[r] = df_b2[M1];
    w3aDr[r] = dr_w3[M0]; w3bDr[r] = dr_w3[M1];
    w3aDf[r] = df_w3[M0]; w3bDf[r] = df_w3[M1];
  }
  const float b3dr = dr_b3[0], b3df = df_b3[0];

  // ---- scan state
  float iv = init_var[p];
  iv = fminf(fmaxf(iv, 0.005f), 0.2f);
  float lv = fast_log(iv);

  const float* dwrow = brownian + (size_t)p * (size_t)T;  // contiguous per lane

  for (int t = 0; t < T; ++t) {
    const float v = lv + 4.0f;

    // layer 1 (rank-1 update) -> f16 B operands, directly in B-layout
    v16h Bdr, Bdf;
    #pragma unroll
    for (int j = 0; j < 16; ++j) {
      Bdr[j] = (_Float16)fast_tanh(pre_dr[j] + v * w1v_dr[j]);
      Bdf[j] = (_Float16)fast_tanh(pre_df[j] + v * w1v_df[j]);
    }

    // layer 2: 4 independent WMMAs (bias rides in C)
    v8f D0 = wmma_f16(A0dr, Bdr, Cdr0);
    v8f D1 = wmma_f16(A1dr, Bdr, Cdr1);
    v8f E0 = wmma_f16(A0df, Bdf, Cdf0);
    v8f E1 = wmma_f16(A1df, Bdf, Cdf1);

    // layer 3: tanh + in-lane 16-dot with w3 (each lane holds 16 of its path's
    // 32 hidden units), then one half-wave exchange completes the reduction.
    float sdr = 0.0f, sdf = 0.0f;
    #pragma unroll
    for (int r = 0; r < 8; ++r) {
      sdr += fast_tanh(D0[r]) * w3aDr[r];
      sdr += fast_tanh(D1[r]) * w3bDr[r];
      sdf += fast_tanh(E0[r]) * w3aDf[r];
      sdf += fast_tanh(E1[r]) * w3bDf[r];
    }
    sdr += __shfl_xor(sdr, 16, 32);
    sdf += __shfl_xor(sdf, 16, 32);

    const float drift = 3.0f * fast_tanh(sdr + b3dr);
    const float diffu = fast_sigmoid(sdf + b3df) + 0.1f;

    lv = lv + drift * dt + diffu * dwrow[t];
    lv = fminf(fmaxf(lv, -7.0f), -0.5f);

    if (half == 0) out[(size_t)t * (size_t)B + p] = fast_exp(lv);

    if ((t & 31) == 0 && (t + 64) < T)
      __builtin_prefetch(dwrow + t + 64, 0, 0);   // global_prefetch ahead of dW stream
  }
}

extern "C" void kernel_launch(void* const* d_in, const int* in_sizes, int n_in,
                              void* d_out, int out_size, void* d_ws, size_t ws_size,
                              hipStream_t stream) {
  const float* init_var = (const float*)d_in[0];
  const float* signatures = (const float*)d_in[1];
  const float* brownian = (const float*)d_in[2];
  const float* dt_p = (const float*)d_in[3];
  const float* dr_w1 = (const float*)d_in[4];
  const float* dr_b1 = (const float*)d_in[5];
  const float* dr_w2 = (const float*)d_in[6];
  const float* dr_b2 = (const float*)d_in[7];
  const float* dr_w3 = (const float*)d_in[8];
  const float* dr_b3 = (const float*)d_in[9];
  const float* df_w1 = (const float*)d_in[10];
  const float* df_b1 = (const float*)d_in[11];
  const float* df_w2 = (const float*)d_in[12];
  const float* df_b2 = (const float*)d_in[13];
  const float* df_w3 = (const float*)d_in[14];
  const float* df_b3 = (const float*)d_in[15];

  const int B = in_sizes[0];            // 8192
  const int T = in_sizes[2] / B;        // 2048

  const int n_tiles = B / 16;           // one wave32 per 16 paths
  neural_rough_sde_kernel<<<n_tiles, 32, 0, stream>>>(
      init_var, signatures, brownian, dt_p,
      dr_w1, dr_b1, dr_w2, dr_b2, dr_w3, dr_b3,
      df_w1, df_b1, df_w2, df_b2, df_w3, df_b3,
      (float*)d_out, B, T);
}
